// SimpleLSTM_6631429505747
// MI455X (gfx1250) — matrix operand order repaired and checked by reading, not compiled, not run on hardware
//
#include <hip/hip_runtime.h>

// ---------------------------------------------------------------------------
// LSTM (B=8192, T=1024, I=3, H=64) + FC(64->1) + sigmoid, for gfx1250.
//
// v5 changes vs v4:
//  - Group the 3 B-fragment LDS loads of each gate chain ahead of the 6 WMMAs
//    that consume them. v4's allocator single-buffered B (same 8 VGPRs for
//    all 48 fragments), forcing s_wait_dscnt 0 before every WMMA pair and
//    exposing full LDS latency. With 3 fragments resident (~+16 regs, peak
//    ~215 < 256) the waits overlap WMMA execution.
//
// Design: gates = [h(64)|x0 x1 x2|1|0...] @ augmented-W^T via
// v_wmma_f32_16x16x32_bf16; M=32 rows/wave so each B fragment feeds 2 WMMAs;
// wave-private LDS activation tile, no per-step barriers (wave DS ops are
// in-order); native v_tanh_f32 activations; LICM fence keeps invariant
// weight-fragment loads as per-step ds_load_b128 (no spills / MSB churn).
// ---------------------------------------------------------------------------

typedef __bf16 bf16;
typedef bf16  v16bf __attribute__((ext_vector_type(16)));
typedef bf16  v8bf  __attribute__((ext_vector_type(8)));
typedef float v8f   __attribute__((ext_vector_type(8)));

#define B_TOT 8192
#define T_LEN 1024
#define I_DIM 3
#define H_DIM 64
#define G4    256          // 4*H gate columns
#define KAUG  96           // 64 h + 3 x + 1 bias + 28 pad (3 K-chunks of 32)
#define WAVES_PER_BLOCK 2
#define ROWS_PER_WAVE   32
#define ROWS_PER_BLOCK  (WAVES_PER_BLOCK * ROWS_PER_WAVE)

// 8 bf16 at p + 8 bf16 at p+16 (two 16B LDS reads) -> v16bf per the ISA 16-bit A/B
// fragment layout (lo 8 -> v[0..7], hi 8 -> v[8..15]).
__device__ __forceinline__ v16bf load_frag(const bf16* p) {
    v8bf lo = *(const v8bf*)(p);
    v8bf hi = *(const v8bf*)(p + 16);
    v16bf r;
#pragma unroll
    for (int i = 0; i < 8; ++i) { r[i] = lo[i]; r[i + 8] = hi[i]; }
    return r;
}

__device__ __forceinline__ float fast_tanh(float x) {
#if __has_builtin(__builtin_amdgcn_tanhf)
    return __builtin_amdgcn_tanhf(x);            // v_tanh_f32: 1 TRANS op
#elif __has_builtin(__builtin_amdgcn_tanh_f32)
    return __builtin_amdgcn_tanh_f32(x);
#else
    float e = __expf(-2.0f * x);
    return (1.0f - e) * __builtin_amdgcn_rcpf(1.0f + e);
#endif
}
__device__ __forceinline__ float fast_sigmoid(float x) {
    return __builtin_fmaf(0.5f, fast_tanh(0.5f * x), 0.5f);
}

__global__ __launch_bounds__(32 * WAVES_PER_BLOCK)
void lstm_wmma_kernel(const float* __restrict__ x,
                      const float* __restrict__ W_ih,
                      const float* __restrict__ W_hh,
                      const float* __restrict__ b_ih,
                      const float* __restrict__ b_hh,
                      const float* __restrict__ W_fc,
                      const float* __restrict__ b_fc,
                      float* __restrict__ out) {
    // Augmented weights, row n = [W_hh[n][:], W_ih[n][:], b_ih[n]+b_hh[n], 0].
    __shared__ __align__(16) bf16 sW[G4][KAUG];                      // 48 KB
    // Per-wave augmented activations: 32 rows x 96.
    __shared__ __align__(16) bf16 sA[WAVES_PER_BLOCK][ROWS_PER_WAVE][KAUG];

    const int tid = threadIdx.x;

    // ---- one-time staging of augmented weights into LDS (bf16) ----
    for (int idx = tid; idx < G4 * KAUG; idx += blockDim.x) {
        const int row = idx / KAUG, col = idx % KAUG;
        float v;
        if (col < H_DIM)                  v = W_hh[row * H_DIM + col];
        else if (col < H_DIM + I_DIM)     v = W_ih[row * I_DIM + (col - H_DIM)];
        else if (col == H_DIM + I_DIM)    v = b_ih[row] + b_hh[row];
        else                              v = 0.0f;
        sW[row][col] = (bf16)v;
    }
    // ---- init activations: h=0, x staged per step, constant-1 at col 67 ----
    for (int idx = tid; idx < WAVES_PER_BLOCK * ROWS_PER_WAVE * KAUG;
         idx += blockDim.x) {
        const int w = idx / (ROWS_PER_WAVE * KAUG);
        const int rem = idx % (ROWS_PER_WAVE * KAUG);
        const int r = rem / KAUG, c = rem % KAUG;
        sA[w][r][c] = (bf16)((c == H_DIM + I_DIM) ? 1.0f : 0.0f);
    }
    __syncthreads();

    const int wave  = tid >> 5;
    const int lane  = tid & 31;
    const int nlo   = lane & 15;        // D-tile column / A row id
    const int hi    = lane >> 4;        // half-wave select
    const int bbase = blockIdx.x * ROWS_PER_BLOCK + wave * ROWS_PER_WAVE;

    // cell state: [m tile][h-col group n][D vgpr g] = 64 f32/lane = 32x64/wave
    float cst[2][4][8];
#pragma unroll
    for (int m = 0; m < 2; ++m)
#pragma unroll
        for (int n = 0; n < 4; ++n)
#pragma unroll
            for (int g = 0; g < 8; ++g) cst[m][n][g] = 0.0f;

    // every lane owns one batch row's x stream (32 rows per wave)
    const float* xrow = x + (size_t)(bbase + lane) * T_LEN * I_DIM;

    const int akoff = hi ? 8 : 0;    // A: lanes 16-31 hold K+8 (ISA layout)
    const int bkoff = hi ? 8 : 0;    // B: lanes 16-31 hold K+8

#pragma unroll 1
    for (int t = 0; t < T_LEN; ++t) {
        // Stop LICM from hoisting the loop-invariant sW fragment loads out of
        // the loop (that costs 384 VGPRs -> MSB churn or scratch spills).
        asm volatile("" ::: "memory");

        // ---- stage x_t (bf16) into augmented A columns 64..66 ----
        {
            const float* xp = xrow + t * I_DIM;
            sA[wave][lane][H_DIM + 0] = (bf16)xp[0];
            sA[wave][lane][H_DIM + 1] = (bf16)xp[1];
            sA[wave][lane][H_DIM + 2] = (bf16)xp[2];
            __builtin_prefetch(xp + 16 * I_DIM, 0, 0);  // ~16 steps ahead
        }
        // no barrier needed: wave-private LDS tile, DS ops in-order per wave

        // ---- A fragments: 2 M-tiles x 3 K-chunks ----
        v16bf afrag[2][3];
#pragma unroll
        for (int m = 0; m < 2; ++m)
#pragma unroll
            for (int kc = 0; kc < 3; ++kc)
                afrag[m][kc] =
                    load_frag(&sA[wave][16 * m + nlo][kc * 32 + akoff]);

        // ---- 16 N-tiles as 4 groups of (i,f,g,o); each B fragment feeds
        //      both M tiles; 3 fragments kept in flight per gate chain ----
#pragma unroll
        for (int n = 0; n < 4; ++n) {
            v8f acc[2][4];
#pragma unroll
            for (int gsel = 0; gsel < 4; ++gsel) {
                const int wrow = 16 * (n + 4 * gsel) + nlo;
                // issue all 3 fragment loads before consuming any of them
                v16bf b[3];
#pragma unroll
                for (int kc = 0; kc < 3; ++kc)
                    b[kc] = load_frag(&sW[wrow][kc * 32 + bkoff]);
                v8f c0 = {}, c1 = {};
#pragma unroll
                for (int kc = 0; kc < 3; ++kc) {
                    c0 = __builtin_amdgcn_wmma_f32_16x16x32_bf16(
                             false, afrag[0][kc], false, b[kc], (short)0, c0,
                             false, false);
                    c1 = __builtin_amdgcn_wmma_f32_16x16x32_bf16(
                             false, afrag[1][kc], false, b[kc], (short)0, c1,
                             false, false);
                }
                acc[0][gsel] = c0;
                acc[1][gsel] = c1;
            }
            // i/f/g/o tiles share lane/VGPR positions -> per-lane math only
#pragma unroll
            for (int m = 0; m < 2; ++m) {
#pragma unroll
                for (int g = 0; g < 8; ++g) {
                    const float ig = fast_sigmoid(acc[m][0][g]);
                    const float fg = fast_sigmoid(acc[m][1][g]);
                    const float gg = fast_tanh(acc[m][2][g]);
                    const float og = fast_sigmoid(acc[m][3][g]);
                    const float cc = fg * cst[m][n][g] + ig * gg;
                    cst[m][n][g] = cc;
                    const float hh = og * fast_tanh(cc);
                    // writeback h (bf16), D layout: row g+8*hi, col 16n+nlo
                    sA[wave][16 * m + g + 8 * hi][16 * n + nlo] = (bf16)hh;
                }
            }
        }
    }
    __syncthreads();

    // ---- final FC: out[b] = sigmoid(h . W_fc + b_fc) ----
    if (tid < ROWS_PER_BLOCK) {
        const int w = tid / ROWS_PER_WAVE;
        const int r = tid % ROWS_PER_WAVE;
        float acc = b_fc[0];
#pragma unroll
        for (int k = 0; k < H_DIM; ++k)
            acc += (float)sA[w][r][k] * W_fc[k];
        out[blockIdx.x * ROWS_PER_BLOCK + tid] = fast_sigmoid(acc);
    }
}

extern "C" void kernel_launch(void* const* d_in, const int* in_sizes, int n_in,
                              void* d_out, int out_size, void* d_ws,
                              size_t ws_size, hipStream_t stream) {
    (void)in_sizes; (void)n_in; (void)out_size; (void)d_ws; (void)ws_size;
    const float* x    = (const float*)d_in[0];
    const float* W_ih = (const float*)d_in[1];
    const float* W_hh = (const float*)d_in[2];
    const float* b_ih = (const float*)d_in[3];
    const float* b_hh = (const float*)d_in[4];
    const float* W_fc = (const float*)d_in[5];
    const float* b_fc = (const float*)d_in[6];
    float* out = (float*)d_out;

    dim3 grid(B_TOT / ROWS_PER_BLOCK);      // 128 blocks
    dim3 block(32 * WAVES_PER_BLOCK);       // 64 threads = 2 wave32
    hipLaunchKernelGGL(lstm_wmma_kernel, grid, block, 0, stream,
                       x, W_ih, W_hh, b_ih, b_hh, W_fc, b_fc, out);
}
